// TorchscriptUnfoldCorrelation_45414984188034
// MI455X (gfx1250) — matrix operand order repaired and checked by reading, not compiled, not run on hardware
//
#include <hip/hip_runtime.h>
#include <hip/hip_bf16.h>

// -----------------------------------------------------------------------------
// FlowNet correlation (k=1, d=4, s1=s2=1), B=16 C=256 H=W=96, out = [16,81,96,96].
//
// MI455X strategy:
//  * Band-of-matmul formulation: for (y, y+dy), out[dx,x] is the width-9 band of
//    x1_row^T (96x256) * x2_padrow (256x104)  -> v_wmma_f32_16x16x32_bf16.
//  * ~fp32 accuracy via bf16 hi/lo split: D += Ahi*Bhi + Ahi*Blo + Alo*Bhi.
//    (hi = trunc16(f), lo = trunc16(f - hi): representation error <= 2^-16 rel.)
//  * Workgroup = (b, 4-row y-block, 16-px x-tile); 18 waves = 9 dy x 2 N-tiles,
//    4 v8f accumulators per wave (32 VGPRs) -> spill-free.
//  * Fill path: chunk-invariant per-thread assignments (addresses/guards hoisted
//    out of the K loop), v_perm_b32 packs bf16 pairs, ds_store_b64 into LDS.
// -----------------------------------------------------------------------------

typedef __attribute__((ext_vector_type(16))) __bf16 v16bf;
typedef __attribute__((ext_vector_type(8)))  float  v8f;

union FragBF {
  uint4 q[2];
  v16bf v;
};

__device__ __forceinline__ uint4 lds_ld128(const unsigned short* p) {
  return *reinterpret_cast<const uint4*>(p);
}

// Pack high 16 bits (bf16 truncation) of two fp32 words: [lo16]=u0>>16, [hi16]=u1>>16.
__device__ __forceinline__ unsigned pack_hi16(unsigned u0, unsigned u1) {
  return __builtin_amdgcn_perm(u1, u0, 0x07060302u);
}

#define CHS 40   // padded channel stride (halves): 80B = multiple of 16B, bank-friendly

__global__ __launch_bounds__(576) void corr_wmma_kernel(
    const float* __restrict__ x1, const float* __restrict__ x2,
    float* __restrict__ out)
{
  // hi/lo bf16 planes. A: 4 rows x 16 px x 32 ch. B: 12 rows x 32 px x 32 ch.
  __shared__ __align__(16) unsigned short sA[2][4][16 * CHS];   // 10.0 KB
  __shared__ __align__(16) unsigned short sB[2][12 * 32 * CHS]; // 60.0 KB
  __shared__ float sStage[9][144];                              //  5.1 KB

  const int tid   = threadIdx.x;
  const int wave  = tid >> 5;      // 0..17
  const int dyi   = wave >> 1;     // 0..8 == dy+4
  const int nt    = wave & 1;      // which 16-col N-tile of the 32-px window
  const int lane  = tid & 31;
  const int laneN = lane & 15;
  const int half  = lane >> 4;

  int bid = blockIdx.x;
  const int t  = bid % 6;  bid /= 6;    // x tile (16 px)
  const int yb = bid % 24;              // y block (4 rows)
  const int b  = bid / 24;
  const int y0 = yb * 4;
  const int x0 = t * 16;

  const int planeHW = 96 * 96;
  const float* x1b = x1 + (size_t)b * 256 * planeHW;
  const float* x2b = x2 + (size_t)b * 256 * planeHW;

  // ---- chunk-invariant fill assignments --------------------------------------
  // threads [0,384):   x2 fill, one (row,px) each: r = tid/32 (0..11), p = tid%32
  // threads [384,448): x1 fill, one (row,px) each: yl = w/16, m = w%16
  const bool isX2 = tid < 384;
  const bool isX1 = (tid >= 384) && (tid < 448);

  const int f_r  = tid >> 5;
  const int f_p  = tid & 31;
  const int f_gy = y0 - 4 + f_r;                 // global row (x2, zero-padded)
  const int f_gx = x0 - 4 + f_p;                 // global col (x2, zero-padded)
  const bool f_in = ((unsigned)f_gy < 96u) && ((unsigned)f_gx < 96u);
  const bool f_ok = isX2 && f_in;
  const float* f_g = x2b + (size_t)f_gy * 96 + f_gx;
  unsigned short* f_l0 = &sB[0][(f_r * 32 + f_p) * CHS];
  unsigned short* f_l1 = &sB[1][(f_r * 32 + f_p) * CHS];

  const int w    = tid - 384;
  const int g_yl = (w >> 4) & 3;
  const int g_m  = w & 15;
  const float* g_g = x1b + (size_t)(y0 + g_yl) * 96 + (x0 + g_m);
  unsigned short* g_l0 = &sA[0][g_yl][g_m * CHS];
  unsigned short* g_l1 = &sA[1][g_yl][g_m * CHS];

  // Zero the padded-border x2 regions once (they are never overwritten).
  if (isX2 && !f_in) {
    uint2 z; z.x = 0u; z.y = 0u;
#pragma unroll
    for (int cg = 0; cg < 8; ++cg) {
      ((uint2*)f_l0)[cg] = z;
      ((uint2*)f_l1)[cg] = z;
    }
  }

  v8f acc[4];
#pragma unroll
  for (int yl = 0; yl < 4; ++yl) acc[yl] = (v8f){};

  for (int kc = 0; kc < 256; kc += 32) {
    // ---- stage x2: 32 ch for this thread's (row,px); immediates index channels
    if (f_ok) {
      const float* gp = f_g + (size_t)kc * planeHW;
#pragma unroll
      for (int cg = 0; cg < 8; ++cg) {
        unsigned u0 = __float_as_uint(gp[(cg * 4 + 0) * planeHW]);
        unsigned u1 = __float_as_uint(gp[(cg * 4 + 1) * planeHW]);
        unsigned u2 = __float_as_uint(gp[(cg * 4 + 2) * planeHW]);
        unsigned u3 = __float_as_uint(gp[(cg * 4 + 3) * planeHW]);
        unsigned l0 = __float_as_uint(__uint_as_float(u0) - __uint_as_float(u0 & 0xffff0000u));
        unsigned l1 = __float_as_uint(__uint_as_float(u1) - __uint_as_float(u1 & 0xffff0000u));
        unsigned l2 = __float_as_uint(__uint_as_float(u2) - __uint_as_float(u2 & 0xffff0000u));
        unsigned l3 = __float_as_uint(__uint_as_float(u3) - __uint_as_float(u3 & 0xffff0000u));
        uint2 hi; hi.x = pack_hi16(u0, u1); hi.y = pack_hi16(u2, u3);
        uint2 lo; lo.x = pack_hi16(l0, l1); lo.y = pack_hi16(l2, l3);
        ((uint2*)f_l0)[cg] = hi;
        ((uint2*)f_l1)[cg] = lo;
      }
    }
    // ---- stage x1: 32 ch for this thread's (row,px)
    if (isX1) {
      const float* gp = g_g + (size_t)kc * planeHW;
#pragma unroll
      for (int cg = 0; cg < 8; ++cg) {
        unsigned u0 = __float_as_uint(gp[(cg * 4 + 0) * planeHW]);
        unsigned u1 = __float_as_uint(gp[(cg * 4 + 1) * planeHW]);
        unsigned u2 = __float_as_uint(gp[(cg * 4 + 2) * planeHW]);
        unsigned u3 = __float_as_uint(gp[(cg * 4 + 3) * planeHW]);
        unsigned l0 = __float_as_uint(__uint_as_float(u0) - __uint_as_float(u0 & 0xffff0000u));
        unsigned l1 = __float_as_uint(__uint_as_float(u1) - __uint_as_float(u1 & 0xffff0000u));
        unsigned l2 = __float_as_uint(__uint_as_float(u2) - __uint_as_float(u2 & 0xffff0000u));
        unsigned l3 = __float_as_uint(__uint_as_float(u3) - __uint_as_float(u3 & 0xffff0000u));
        uint2 hi; hi.x = pack_hi16(u0, u1); hi.y = pack_hi16(u2, u3);
        uint2 lo; lo.x = pack_hi16(l0, l1); lo.y = pack_hi16(l2, l3);
        ((uint2*)g_l0)[cg] = hi;
        ((uint2*)g_l1)[cg] = lo;
      }
    }
    __syncthreads();

    // ---- WMMA: per wave, 4 y-rows x (its one N-tile), 3 bf16 matmuls each ----
#pragma unroll
    for (int yl = 0; yl < 4; ++yl) {
      // A fragment (16-bit 16x32 layout): lane L: M=L%16, Kbase=(L/16)*8;
      // elems 0..7 = K [Kbase, Kbase+8), elems 8..15 = K [16+Kbase, 16+Kbase+8)
      FragBF ahi, alo;
      const unsigned short* a0 = &sA[0][yl][laneN * CHS + half * 8];
      const unsigned short* a1 = &sA[1][yl][laneN * CHS + half * 8];
      ahi.q[0] = lds_ld128(a0);
      ahi.q[1] = lds_ld128(a0 + 16);
      alo.q[0] = lds_ld128(a1);
      alo.q[1] = lds_ld128(a1 + 16);

      // B fragment (16-bit 32x16): lane L: N=L%16, Kbase=(L/16)*8; elems 0..7 = K [Kbase,Kbase+8), 8..15 = K [16+Kbase,16+Kbase+8)
      const int r = yl + dyi;                 // sB row, 0..11
      const int p = nt * 16 + laneN;          // padded column within 32-px window
      const unsigned short* b0 = &sB[0][(r * 32 + p) * CHS + half * 8];
      const unsigned short* b1 = &sB[1][(r * 32 + p) * CHS + half * 8];
      FragBF bhi, blo;
      bhi.q[0] = lds_ld128(b0);
      bhi.q[1] = lds_ld128(b0 + 16);
      blo.q[0] = lds_ld128(b1);
      blo.q[1] = lds_ld128(b1 + 16);

      v8f c = acc[yl];
      c = __builtin_amdgcn_wmma_f32_16x16x32_bf16(false, ahi.v, false, bhi.v,
                                                  (short)0, c, false, false);
      c = __builtin_amdgcn_wmma_f32_16x16x32_bf16(false, ahi.v, false, blo.v,
                                                  (short)0, c, false, false);
      c = __builtin_amdgcn_wmma_f32_16x16x32_bf16(false, alo.v, false, bhi.v,
                                                  (short)0, c, false, false);
      acc[yl] = c;
    }
    __syncthreads();
  }

  // ---- write-out: band extraction via LDS staging, coalesced stores ----
  // C/D layout: VGPR v holds M = v + 8*(lane/16), N = lane%16.
  // N-tile 0: dx+4 = n - m (0..8).  N-tile 1: dx+4 = n - m + 16 (0..8).
  // For a given (m, dx+4) exactly one of the two nt-waves produces the value.
  const float norm = 1.0f / 256.0f;
#pragma unroll 1
  for (int yl = 0; yl < 4; ++yl) {
    float* stage = sStage[dyi];
#pragma unroll
    for (int v = 0; v < 8; ++v) {
      const int m = v + half * 8;
      const int n = laneN;
      const int d = n - m + nt * 16;          // dx+4 candidate
      if (d >= 0 && d <= 8) stage[d * 16 + m] = acc[yl][v];
    }
    __syncthreads();
    const int y = y0 + yl;
    for (int idx = tid; idx < 1296; idx += 576) {   // 9 dy regions x 144 floats
      const int dyo = idx / 144;
      const int e   = idx - dyo * 144;
      const int dx4 = e >> 4;
      const int m   = e & 15;
      const int i   = dyo * 9 + dx4;                // (dy+4)*9 + (dx+4)
      out[(((size_t)b * 81 + i) * 96 + y) * 96 + (x0 + m)] = sStage[dyo][e] * norm;
    }
    __syncthreads();  // staging reused by next yl
  }
}

extern "C" void kernel_launch(void* const* d_in, const int* in_sizes, int n_in,
                              void* d_out, int out_size, void* d_ws, size_t ws_size,
                              hipStream_t stream) {
  (void)in_sizes; (void)n_in; (void)d_ws; (void)ws_size; (void)out_size;
  const float* x1 = (const float*)d_in[0];
  const float* x2 = (const float*)d_in[1];
  float* out = (float*)d_out;
  // grid: b (16) x y-blocks (24) x x-tiles (6) = 2304 WGs; 18 waves (576 threads) each.
  dim3 grid(16 * 24 * 6);
  dim3 block(576);
  hipLaunchKernelGGL(corr_wmma_kernel, grid, block, 0, stream, x1, x2, out);
}
